// PesudoLabelGenerator_v9_8_rev_s_3702261809695
// MI455X (gfx1250) — matrix operand order repaired and checked by reading, not compiled, not run on hardware
//
#include <hip/hip_runtime.h>
#include <math.h>

#define NPTS 8192
#define MPTS 8192
#define CDIM 96
#define NOUT 20
#define SPLIT 4
#define NSPAN (NPTS / SPLIT)   // 2048
#define MSPAN (MPTS / SPLIT)   // 2048

// d_out layout (floats, concatenated in reference return order)
#define O_TRUST  0
#define O_MF     8192                    // N*C
#define O_COORDS (O_MF + 786432)         // N*3
#define O_PRE    (O_COORDS + 24576)      // N*20
#define O_UW     (O_PRE + 163840)        // N*20
#define O_SVU    (O_UW + 163840)         // N*20
#define O_SMI    (O_SVU + 163840)        // M

typedef __attribute__((ext_vector_type(16))) __bf16   v16bf;
typedef __attribute__((ext_vector_type(8)))  float    v8f;
typedef __attribute__((ext_vector_type(4)))  unsigned int v4u;

union Frag { v4u q[2]; v16bf v; };

__device__ __forceinline__ unsigned short f2bf(float x) {
  unsigned int u = __float_as_uint(x);
  u += 0x7fffu + ((u >> 16) & 1u);
  return (unsigned short)(u >> 16);
}

// ---------------- A0: 4x4 inverse + diff matrix, reset flags ----------------
__global__ void plg_kA0(const float* __restrict__ posses, float* __restrict__ diffm,
                        int* __restrict__ flags) {
  if (threadIdx.x == 0 && blockIdx.x == 0) {
    flags[0] = 0; flags[1] = 0;
    float a[4][8];
    for (int i = 0; i < 4; ++i)
      for (int j = 0; j < 4; ++j) { a[i][j] = posses[16 + i*4 + j]; a[i][4+j] = (i==j) ? 1.f : 0.f; }
    for (int col = 0; col < 4; ++col) {
      int p = col; float best = fabsf(a[col][col]);
      for (int r = col+1; r < 4; ++r) { float v = fabsf(a[r][col]); if (v > best) { best = v; p = r; } }
      if (p != col) for (int j = 0; j < 8; ++j) { float t = a[col][j]; a[col][j] = a[p][j]; a[p][j] = t; }
      float d = 1.f / a[col][col];
      for (int j = 0; j < 8; ++j) a[col][j] *= d;
      for (int r = 0; r < 4; ++r) if (r != col) {
        float f = a[r][col];
        for (int j = 0; j < 8; ++j) a[r][j] -= f * a[col][j];
      }
    }
    for (int i = 0; i < 4; ++i)
      for (int j = 0; j < 4; ++j) {
        float s = 0.f;
        for (int k = 0; k < 4; ++k) s += a[i][4+k] * posses[k*4 + j];
        diffm[i*4 + j] = s;
      }
  }
}

// ---------------- A: bf16 conversion, norms, aligned coords, cls/upd --------
__global__ void plg_kA(const float* __restrict__ surF, const float* __restrict__ surC,
                       const float* __restrict__ surGT, const float* __restrict__ meanF,
                       const float* __restrict__ oriC, const int* __restrict__ ranMask,
                       const int* __restrict__ lgp, const float* __restrict__ diffm,
                       unsigned short* __restrict__ meanBf, unsigned short* __restrict__ surBf,
                       float* __restrict__ nmeta, float* __restrict__ mmeta) {
  const int t = blockIdx.x * blockDim.x + threadIdx.x;
  const int lg = *lgp;
  if (t < NPTS) {
    float s2 = 0.f;
    for (int c = 0; c < CDIM; ++c) { float v = meanF[(size_t)t*CDIM + c]; s2 += v*v; meanBf[(size_t)t*CDIM + c] = f2bf(v); }
    float inv = 1.f / fmaxf(sqrtf(s2), 1e-4f);
    float cx = oriC[t*3+0], cy = oriC[t*3+1], cz = oriC[t*3+2];
    float* nm = nmeta + (size_t)t * 8;
    nm[0]=inv; nm[1]=cx*cx+cy*cy+cz*cz; nm[2]=cx; nm[3]=cy; nm[4]=cz; nm[5]=s2; nm[6]=0.f; nm[7]=0.f;
  }
  if (t < MPTS) {
    float s2 = 0.f;
    for (int c = 0; c < CDIM; ++c) { float v = surF[(size_t)t*CDIM + c]; s2 += v*v; surBf[(size_t)t*CDIM + c] = f2bf(v); }
    float inv = 1.f / fmaxf(sqrtf(s2), 1e-4f);
    float hx = surC[t*3+0], hy = surC[t*3+1], hz = surC[t*3+2];
    float ax = diffm[0]*hx + diffm[1]*hy + diffm[2]*hz  + diffm[3];
    float ay = diffm[4]*hx + diffm[5]*hy + diffm[6]*hz  + diffm[7];
    float az = diffm[8]*hx + diffm[9]*hy + diffm[10]*hz + diffm[11];
    int cls = 0; float bv = surGT[(size_t)t*NOUT];
    for (int c = 1; c < NOUT; ++c) { float v = surGT[(size_t)t*NOUT + c]; if (v > bv) { bv = v; cls = c; } }
    bool upd = (cls < 9) || (t < lg) || (ranMask[t] == 1);
    float* mm = mmeta + (size_t)t * 8;
    mm[0]=inv; mm[1]=ax*ax+ay*ay+az*az; mm[2]=ax; mm[3]=ay; mm[4]=az;
    mm[5]=upd ? 1.f : 0.f; mm[6]=(float)cls; mm[7]=0.f;
  }
}

// ---------------- B: fused bf16-WMMA sim tiles + split argmin over N --------
#define BCHUNK 1024
__device__ __forceinline__ void plg_loadB(const unsigned short* __restrict__ meanBf,
                                          int ncol, int hf, Frag* f) {
  const unsigned short* brow = meanBf + (size_t)ncol * CDIM + hf*8;
  #pragma unroll
  for (int j = 0; j < 3; ++j) {
    f[j].q[0] = *reinterpret_cast<const v4u*>(brow + j*32);
    f[j].q[1] = *reinterpret_cast<const v4u*>(brow + j*32 + 16);
  }
}

__global__ __launch_bounds__(256, 2)
void plg_kB(const unsigned short* __restrict__ meanBf, const unsigned short* __restrict__ surBf,
            const float* __restrict__ nmeta, const float* __restrict__ mmeta,
            float* __restrict__ bpartV, int* __restrict__ bpartI) {
  __shared__ v4u smemq[BCHUNK * 2];   // 32KB: {invF,cn2,cx,cy,cz,nf2,_,_} per n
  const int tid  = threadIdx.x;
  const int lane = tid & 31;
  const int wave = tid >> 5;
  const int lr   = lane & 15;
  const int hf   = lane >> 4;
  const int split = blockIdx.x & (SPLIT - 1);
  const int mblk  = blockIdx.x >> 2;
  const int m_base = mblk * 128 + wave * 16;

  // A fragments: sur_sv_feature rows (loop invariant). Lane holds one m-row's K-slices.
  Frag afr[3];
  {
    const unsigned short* row = surBf + (size_t)(m_base + lr) * CDIM;
    #pragma unroll
    for (int j = 0; j < 3; ++j) {
      afr[j].q[0] = *reinterpret_cast<const v4u*>(row + j*32 +      hf*8);
      afr[j].q[1] = *reinterpret_cast<const v4u*>(row + j*32 + 16 + hf*8);
    }
  }
  // per-lane m-row invariants (D rows = hf*8 + v)
  float invG[8], am2[8], ax[8], ay[8], az[8];
  #pragma unroll
  for (int v = 0; v < 8; ++v) {
    const float* mm = mmeta + (size_t)(m_base + hf*8 + v) * 8;
    invG[v]=mm[0]; am2[v]=mm[1]; ax[v]=mm[2]; ay[v]=mm[3]; az[v]=mm[4];
  }
  float minv[8]; int mini[8];
  #pragma unroll
  for (int v = 0; v < 8; ++v) { minv[v] = 3.4e38f; mini[v] = 0; }

  const int n0 = split * NSPAN;
  for (int cb = n0; cb < n0 + NSPAN; cb += BCHUNK) {
    // issue first B-fragment of this chunk before the staging barrier (overlap)
    Frag cur[3];
    plg_loadB(meanBf, cb + lr, hf, cur);
    __syncthreads();
    for (int r = tid; r < BCHUNK; r += 256) {
      const v4u* src = reinterpret_cast<const v4u*>(nmeta + (size_t)(cb + r) * 8);
      smemq[r*2 + 0] = src[0];
      smemq[r*2 + 1] = src[1];
    }
    __syncthreads();

    auto tile = [&](int nb, const Frag* bf) {
      const int ncol = cb + nb + lr;
      v8f acc = {0.f,0.f,0.f,0.f,0.f,0.f,0.f,0.f};
      #pragma unroll
      for (int j = 0; j < 3; ++j)
        acc = __builtin_amdgcn_wmma_f32_16x16x32_bf16(false, afr[j].v, false, bf[j].v,
                                                      (short)0, acc, false, false);
      const float* nm = reinterpret_cast<const float*>(smemq + (size_t)(nb + lr) * 2);
      const float invF = nm[0], cn2 = nm[1], cx = nm[2], cy = nm[3], cz = nm[4];
      #pragma unroll
      for (int v = 0; v < 8; ++v) {
        float dot = acc[v];
        float d = cn2 + am2[v] - 2.f * (ax[v]*cx + ay[v]*cy + az[v]*cz);
        d = fmaxf(d, 0.f);
        float sim = 2.f - dot * invF * invG[v] - __expf(-2.f * d);
        if (sim < minv[v]) { minv[v] = sim; mini[v] = ncol; }
      }
    };

    // software pipeline with peeled epilogue: buffer rotation is unconditional
    for (int nb = 0; nb < BCHUNK - 16; nb += 16) {
      Frag nxt[3];
      plg_loadB(meanBf, cb + nb + 16 + lr, hf, nxt);
      if (nb + 32 < BCHUNK)
        __builtin_prefetch(meanBf + (size_t)(cb + nb + 32 + lr) * CDIM, 0, 1);
      tile(nb, cur);
      cur[0] = nxt[0]; cur[1] = nxt[1]; cur[2] = nxt[2];
    }
    tile(BCHUNK - 16, cur);
  }
  // reduce (min,idx) across the 16 lanes of each half (wave32 halves)
  #pragma unroll
  for (int v = 0; v < 8; ++v) {
    float mv = minv[v]; int mi = mini[v];
    #pragma unroll
    for (int mb = 1; mb < 16; mb <<= 1) {
      float ov = __shfl_xor(mv, mb, 32);
      int   oi = __shfl_xor(mi, mb, 32);
      if (ov < mv || (ov == mv && oi < mi)) { mv = ov; mi = oi; }
    }
    minv[v] = mv; mini[v] = mi;
  }
  if (lr == 0) {
    #pragma unroll
    for (int v = 0; v < 8; ++v) {
      const int row = m_base + hf*8 + v;
      bpartV[(size_t)split * MPTS + row] = minv[v];
      bpartI[(size_t)split * MPTS + row] = mini[v];
    }
  }
}

// ---------------- Br: merge split argmins (ascending n keeps first-min) -----
__global__ void plg_kBr(const float* __restrict__ bpartV, const int* __restrict__ bpartI,
                        int* __restrict__ smidx) {
  const int t = blockIdx.x * blockDim.x + threadIdx.x;
  if (t >= MPTS) return;
  float bv = bpartV[t]; int bi = bpartI[t];
  #pragma unroll
  for (int s = 1; s < SPLIT; ++s) {
    float v = bpartV[(size_t)s * MPTS + t];
    int   i = bpartI[(size_t)s * MPTS + t];
    if (v < bv) { bv = v; bi = i; }
  }
  smidx[t] = bi;
}

// ---------------- B2: gather mid metadata via sm_index ----------------------
__global__ void plg_kB2(const int* __restrict__ smidx, const float* __restrict__ nmeta,
                        const float* __restrict__ mmeta, float* __restrict__ midmeta) {
  const int t = blockIdx.x * blockDim.x + threadIdx.x;
  if (t >= MPTS) return;
  const int s = smidx[t];
  const float* nm = nmeta + (size_t)s * 8;
  const float* mm = mmeta + (size_t)t * 8;
  float* md = midmeta + (size_t)t * 8;
  md[0] = nm[5];              // |mid_feat|^2
  md[1] = nm[2]; md[2] = nm[3]; md[3] = nm[4];   // mid coords
  md[4] = mm[6];              // class
  md[5] = mm[5];              // upd
  md[6] = nm[1];              // |mid_coord|^2
  md[7] = 0.f;
}

// ---------------- C: fused bf16-WMMA dist tiles -> split class partials -----
__device__ __forceinline__ void plg_loadA(const unsigned short* __restrict__ meanBf,
                                          int srow, int hf, Frag* f) {
  const unsigned short* arow = meanBf + (size_t)srow * CDIM;
  #pragma unroll
  for (int j = 0; j < 3; ++j) {
    f[j].q[0] = *reinterpret_cast<const v4u*>(arow + j*32 +      hf*8);
    f[j].q[1] = *reinterpret_cast<const v4u*>(arow + j*32 + 16 + hf*8);
  }
}

__global__ __launch_bounds__(256, 2)
void plg_kC(const unsigned short* __restrict__ meanBf, const int* __restrict__ smidx,
            const float* __restrict__ nmeta, const float* __restrict__ midmeta,
            float* __restrict__ cpart) {
  const int tid  = threadIdx.x;
  const int lane = tid & 31;
  const int wave = tid >> 5;
  const int lr   = lane & 15;
  const int hf   = lane >> 4;
  const int split = blockIdx.x & (SPLIT - 1);
  const int nblk  = blockIdx.x >> 2;
  const int n_base = nblk * 128 + wave * 16;
  const int ncol = n_base + lr;
  const int m0 = split * MSPAN;
  const int m1 = m0 + MSPAN;

  // B fragments: mean_features column (loop invariant)
  Frag bfr[3];
  {
    const unsigned short* row = meanBf + (size_t)ncol * CDIM + hf*8;
    #pragma unroll
    for (int j = 0; j < 3; ++j) {
      bfr[j].q[0] = *reinterpret_cast<const v4u*>(row + j*32);
      bfr[j].q[1] = *reinterpret_cast<const v4u*>(row + j*32 + 16);
    }
  }
  const float* nmp = nmeta + (size_t)ncol * 8;
  const float nf2 = nmp[5], cn2 = nmp[1], cx = nmp[2], cy = nmp[3], cz = nmp[4];

  float accv[21];
  #pragma unroll
  for (int c = 0; c < 21; ++c) accv[c] = 0.f;

  auto tileC = [&](int mb, const Frag* a) {
    v8f d = {0.f,0.f,0.f,0.f,0.f,0.f,0.f,0.f};
    #pragma unroll
    for (int j = 0; j < 3; ++j)
      d = __builtin_amdgcn_wmma_f32_16x16x32_bf16(false, a[j].v, false, bfr[j].v,
                                                  (short)0, d, false, false);
    #pragma unroll
    for (int v = 0; v < 8; ++v) {
      const float* md = midmeta + (size_t)(mb + hf*8 + v) * 8;
      float dot = d[v];
      float l2f = fmaxf(nf2 + md[0] - 2.f * dot, 0.f);
      float l2c = fmaxf(cn2 + md[6] - 2.f * (cx*md[1] + cy*md[2] + cz*md[3]), 0.f);
      float w = __expf(-8.f * l2c - 5.5555553f * l2f) * md[5];
      int ci = (int)md[4];
      #pragma unroll
      for (int c = 0; c < NOUT; ++c) accv[c] += (ci == c) ? w : 0.f;
      accv[20] += w;
    }
  };

  // prologue: gather first A fragment; peeled epilogue keeps rotation unconditional
  Frag afr[3];
  plg_loadA(meanBf, smidx[m0 + lr], hf, afr);
  for (int mb = m0; mb < m1 - 16; mb += 16) {
    const int s_nxt = smidx[mb + 16 + lr];   // issue gather index early
    Frag nxt[3];
    plg_loadA(meanBf, s_nxt, hf, nxt);       // overlaps WMMA + VALU tail below
    tileC(mb, afr);
    afr[0] = nxt[0]; afr[1] = nxt[1]; afr[2] = nxt[2];
  }
  tileC(m1 - 16, afr);

  #pragma unroll
  for (int c = 0; c < 21; ++c) accv[c] += __shfl_xor(accv[c], 16, 32);
  if (lane < 16) {
    float* dst = cpart + ((size_t)split * NPTS + ncol) * 24;
    #pragma unroll
    for (int c = 0; c < 21; ++c) dst[c] = accv[c];
  }
}

// ---------------- Cr: merge split partials, softmax, uw/svu -----------------
__global__ void plg_kCr(const float* __restrict__ cpart, const float* __restrict__ svProb,
                        float* __restrict__ outUW, float* __restrict__ outSVU) {
  const int t = blockIdx.x * blockDim.x + threadIdx.x;
  if (t >= NPTS) return;
  float acc[21];
  #pragma unroll
  for (int c = 0; c < 21; ++c) acc[c] = 0.f;
  for (int s = 0; s < SPLIT; ++s) {
    const float* src = cpart + ((size_t)s * NPTS + t) * 24;
    #pragma unroll
    for (int c = 0; c < 21; ++c) acc[c] += src[c];
  }
  float denom = acc[20] + 1e-16f;
  float p[NOUT]; float mx = -3.4e38f;
  for (int c = 0; c < NOUT; ++c) { p[c] = svProb[(size_t)t*NOUT + c]; mx = fmaxf(mx, p[c]); }
  float sum = 0.f;
  for (int c = 0; c < NOUT; ++c) { p[c] = __expf(p[c] - mx); sum += p[c]; }
  float isum = 1.f / sum;
  for (int c = 0; c < NOUT; ++c) {
    float uw = acc[c] / denom;
    outUW [(size_t)t*NOUT + c] = uw;
    outSVU[(size_t)t*NOUT + c] = 0.5f * p[c] * isum + 0.5f * uw;
  }
}

// ---------------- D0..D3: scatter update + trust/pre finalize ---------------
__global__ void plg_kD0(const int* __restrict__ smidx, const float* __restrict__ mmeta,
                        const float* __restrict__ svu, const int* __restrict__ lgp,
                        float* __restrict__ vals, int* __restrict__ flags) {
  const int t = blockIdx.x * blockDim.x + threadIdx.x;
  if (t >= *lgp) return;
  const int s = smidx[t];
  const int ci = (int)mmeta[(size_t)t*8 + 6];
  const float v = svu[(size_t)s*NOUT + ci];
  vals[t] = v;
  if (v > 0.1f) atomicOr(&flags[0], 1);
}

__global__ void plg_kD1(const int* __restrict__ smidx, const float* __restrict__ mmeta,
                        const float* __restrict__ vals, const int* __restrict__ lgp,
                        const int* __restrict__ flags, float* __restrict__ svu) {
  const int t = blockIdx.x * blockDim.x + threadIdx.x;
  if (t >= *lgp) return;
  const bool any1 = (flags[0] != 0);
  const float v = vals[t];
  const bool tmp = any1 ? (v > 0.1f) : (v > 0.0f);
  if (tmp) {
    const int s = smidx[t];
    const int ci = (int)mmeta[(size_t)t*8 + 6];
    for (int k = 0; k < NOUT; ++k) svu[(size_t)s*NOUT + k] = (k == ci) ? 1.f : 0.f;
  }
}

__global__ void plg_kD2(const float* __restrict__ svu, float* __restrict__ maxv,
                        int* __restrict__ flags) {
  const int t = blockIdx.x * blockDim.x + threadIdx.x;
  if (t >= NPTS) return;
  float mv = svu[(size_t)t*NOUT];
  for (int c = 1; c < NOUT; ++c) mv = fmaxf(mv, svu[(size_t)t*NOUT + c]);
  maxv[t] = mv;
  if (mv >= 0.9f) atomicOr(&flags[1], 1);
}

__global__ void plg_kD3(const float* __restrict__ meanF, const float* __restrict__ oriC,
                        const float* __restrict__ svu, const float* __restrict__ maxv,
                        const int* __restrict__ flags, const int* __restrict__ smidx,
                        float* __restrict__ out) {
  const int t = blockIdx.x * blockDim.x + threadIdx.x;
  if (t < NPTS) {
    const float mv = maxv[t];
    const bool tr = (flags[1] != 0) ? (mv >= 0.9f) : (mv >= 0.85f);
    const float mm = tr ? 1.f : 0.f;
    out[O_TRUST + t] = mm;
    for (int c = 0; c < CDIM; ++c) out[O_MF + (size_t)t*CDIM + c] = meanF[(size_t)t*CDIM + c] * mm;
    for (int c = 0; c < 3; ++c)    out[O_COORDS + t*3 + c]        = oriC[t*3 + c] * mm;
    int bi = 0; float bv = svu[(size_t)t*NOUT];
    for (int c = 1; c < NOUT; ++c) { float v = svu[(size_t)t*NOUT + c]; if (v > bv) { bv = v; bi = c; } }
    for (int c = 0; c < NOUT; ++c) out[O_PRE + (size_t)t*NOUT + c] = ((c == bi) ? 1.f : 0.f) * mm;
  }
  if (t < MPTS) out[O_SMI + t] = (float)smidx[t];
}

// ---------------- launch ----------------------------------------------------
extern "C" void kernel_launch(void* const* d_in, const int* in_sizes, int n_in,
                              void* d_out, int out_size, void* d_ws, size_t ws_size,
                              hipStream_t stream) {
  const float* surF   = (const float*)d_in[0];
  const float* surC   = (const float*)d_in[1];
  const float* surGT  = (const float*)d_in[2];
  const float* svProb = (const float*)d_in[3];
  const float* meanF  = (const float*)d_in[4];
  const float* oriC   = (const float*)d_in[5];
  const float* posses = (const float*)d_in[6];
  const int*   ranMask= (const int*)d_in[7];
  const int*   lgp    = (const int*)d_in[8];

  char* ws = (char*)d_ws;
  float*          diffm   = (float*)(ws + 0);                                   //   64 B
  int*            flags   = (int*)(ws + 64);                                    //    8 B
  float*          nmeta   = (float*)(ws + 256);                                 //  256 KB
  float*          mmeta   = (float*)(ws + 256 + 262144);                        //  256 KB
  float*          midmeta = (float*)(ws + 256 + 2*262144);                      //  256 KB
  unsigned short* meanBf  = (unsigned short*)(ws + 256 + 3*262144);             //  1.5 MB
  unsigned short* surBf   = (unsigned short*)(ws + 256 + 3*262144 + 1572864);   //  1.5 MB
  int*            smidx   = (int*)(ws + 256 + 3*262144 + 2*1572864);            //   32 KB
  float*          vals    = (float*)(ws + 256 + 3*262144 + 2*1572864 + 32768);  //    8 KB
  float*          maxv    = (float*)(ws + 256 + 3*262144 + 2*1572864 + 40960);  //   32 KB
  float*          bpartV  = (float*)(ws + 256 + 3*262144 + 2*1572864 + 73728);  //  128 KB
  int*            bpartI  = (int*)  (ws + 256 + 3*262144 + 2*1572864 + 204800); //  128 KB
  float*          cpart   = (float*)(ws + 256 + 3*262144 + 2*1572864 + 335872); //    3 MB

  float* out    = (float*)d_out;
  float* outUW  = out + O_UW;
  float* outSVU = out + O_SVU;

  plg_kA0<<<1, 1, 0, stream>>>(posses, diffm, flags);
  plg_kA <<<32, 256, 0, stream>>>(surF, surC, surGT, meanF, oriC, ranMask, lgp, diffm,
                                  meanBf, surBf, nmeta, mmeta);
  plg_kB <<<(MPTS/128)*SPLIT, 256, 0, stream>>>(meanBf, surBf, nmeta, mmeta, bpartV, bpartI);
  plg_kBr<<<32, 256, 0, stream>>>(bpartV, bpartI, smidx);
  plg_kB2<<<32, 256, 0, stream>>>(smidx, nmeta, mmeta, midmeta);
  plg_kC <<<(NPTS/128)*SPLIT, 256, 0, stream>>>(meanBf, smidx, nmeta, midmeta, cpart);
  plg_kCr<<<32, 256, 0, stream>>>(cpart, svProb, outUW, outSVU);
  plg_kD0<<<8, 256, 0, stream>>>(smidx, mmeta, outSVU, lgp, vals, flags);
  plg_kD1<<<8, 256, 0, stream>>>(smidx, mmeta, vals, lgp, flags, outSVU);
  plg_kD2<<<32, 256, 0, stream>>>(outSVU, maxv, flags);
  plg_kD3<<<32, 256, 0, stream>>>(meanF, oriC, outSVU, maxv, flags, smidx, out);
}
